// PNA_Net_1992864825524
// MI455X (gfx1250) — compile-verified, matrix operand order checked by reading, not run on hardware
//
#include <hip/hip_runtime.h>
#include <hip/hip_bf16.h>
#include <math.h>

// ---------------- problem constants ----------------
#define NN    16384
#define EE    262144
#define BB    64
#define FF    32
#define FEE   16
#define TT    8
#define FOUTT 4
#define LL    3
#define HH    128
#define EPSF  1e-5f

#define EPB   64          // edges per block in the WMMA edge kernel
#define MROWP 104         // padded LDS row stride (bf16 elems): 52 dwords -> conflict-free

typedef __attribute__((ext_vector_type(16))) __bf16 v16bf;
typedef __attribute__((ext_vector_type(8)))  float  v8f;

// ---------------- helpers ----------------
__device__ __forceinline__ unsigned short f2bf(float f) {
  unsigned u = __builtin_bit_cast(unsigned, f);
  unsigned r = u + 0x7FFFu + ((u >> 16) & 1u);   // round-to-nearest-even
  return (unsigned short)(r >> 16);
}
// order-preserving float<->uint map so uint atomicMin/Max == float min/max
__device__ __forceinline__ unsigned encf(float f) {
  unsigned u = __builtin_bit_cast(unsigned, f);
  return (u & 0x80000000u) ? ~u : (u | 0x80000000u);
}
__device__ __forceinline__ float decf(unsigned u) {
  unsigned b = (u & 0x80000000u) ? (u & 0x7FFFFFFFu) : ~u;
  return __builtin_bit_cast(float, b);
}
__device__ __forceinline__ unsigned pk2(float a, float b) {
  return (unsigned)f2bf(a) | ((unsigned)f2bf(b) << 16);
}

// ---------------- generic fill ----------------
__global__ void fill_u32(unsigned* __restrict__ p, unsigned v, long long count) {
  long long i = (long long)blockIdx.x * blockDim.x + threadIdx.x;
  if (i < count) p[i] = v;
}
// init interleaved aggregate planes: {sum=0, sq=0, minEnc=0xFFFFFFFF, maxEnc=0}
__global__ void fill_agg4(uint4* __restrict__ p, long long count) {
  long long i = (long long)blockIdx.x * blockDim.x + threadIdx.x;
  if (i < count) p[i] = make_uint4(0u, 0u, 0xFFFFFFFFu, 0u);
}

// ---------------- node embedding gather: h = node_emb[x] ----------------
__global__ void embed_kernel(const int* __restrict__ x, const float* __restrict__ emb,
                             float* __restrict__ h) {
  int i = blockIdx.x * blockDim.x + threadIdx.x;     // N*32
  if (i < NN * FF) h[i] = emb[x[i >> 5] * FF + (i & 31)];
}

// ---------------- degree + scalers ----------------
__global__ void deg_count_kernel(const int* __restrict__ edge_index, float* __restrict__ deg) {
  int e = blockIdx.x * blockDim.x + threadIdx.x;
  if (e < EE) atomicAdd(&deg[edge_index[EE + e]], 1.0f);
}
__global__ void log_sum_kernel(const float* __restrict__ deg, float* __restrict__ acc) {
  int n = blockIdx.x * blockDim.x + threadIdx.x;
  if (n < NN) atomicAdd(acc, logf(deg[n] + 1.0f));
}
__global__ void scalers_kernel(const float* __restrict__ deg, const float* __restrict__ acc,
                               float* __restrict__ ampA, float* __restrict__ attA) {
  int n = blockIdx.x * blockDim.x + threadIdx.x;
  if (n >= NN) return;
  float avg = acc[0] / (float)NN;
  float dc  = fmaxf(deg[n], 1.0f);
  float ld  = logf(dc + 1.0f);
  ampA[n] = ld / avg;
  attA[n] = avg / ld;
}

// ---------------- repack Wpre[l,t,96,32] -> WMMA B panels (bf16) ----------------
// panel = ntile*3 + kc   (ntile 0..15 over 256 output cols, kc 0..2 over K=96)
// within panel: [lane 0..31][elem 0..15] so a lane's fragment is 32 contiguous bytes.
// B layout (16-bit 32x16, wave32): lane l -> N = l%16 ; elem e -> K = (e%8) + 16*(e/8) + 8*(l/16)
__global__ void repack_kernel(const float* __restrict__ Wpre, unsigned short* __restrict__ Wpk) {
  int idx = blockIdx.x * blockDim.x + threadIdx.x;   // 3*48*512
  if (idx >= LL * 48 * 512) return;
  int l     = idx / (48 * 512);
  int rem   = idx % (48 * 512);
  int panel = rem / 512;
  int pe    = rem % 512;
  int ntile = panel / 3, kc = panel % 3;
  int lane  = pe / 16,  e  = pe % 16;
  int ncol  = ntile * 16 + (lane & 15);
  int K     = kc * 32 + (e & 7) + 16 * (e >> 3) + 8 * (lane >> 4);
  int t = ncol >> 5, f = ncol & 31;
  float w = Wpre[((l * TT + t) * (3 * FF) + K) * FF + f];
  Wpk[idx] = f2bf(w);
}

// ---------------- edge message GEMM (WMMA) + segment aggregation ----------------
// per block: 64 edges; msg[64,256] = [h_dst|h_src|e][64,96] @ Wcat[96,256]  (bf16 in, f32 acc)
// 8 waves: wave covers 2 N-tiles; B fragments (3 K-chunks) held in registers and reused
// across 4 M-tiles of 16 edges each -> 24 WMMAs per wave.
// Aggregates are interleaved per element: agg4[(node*256+col)*4 + {0:sum,1:sq,2:minE,3:maxE}]
// so the atomic burst needs one 64-bit address + immediate offsets.
__global__ __launch_bounds__(256) void edge_msg_wmma_kernel(
    const int* __restrict__ edge_index, const int* __restrict__ edge_attr,
    const float* __restrict__ edge_emb, const float* __restrict__ Weel,
    const float* __restrict__ beel, const float* __restrict__ bprel,
    const float* __restrict__ h, const unsigned short* __restrict__ Wpk,
    unsigned* __restrict__ agg4) {
  __shared__ __align__(16) unsigned short mt[EPB][MROWP];  // bf16 bits, 208B row stride
  __shared__ unsigned sdstScaled[EPB];                     // dst * 1024 (element offset)

  const int tid = threadIdx.x;
  const int e0  = blockIdx.x * EPB;

  // ---- stage A: build m-tile in LDS (4 threads per edge, 8 cols per segment each) ----
  {
    int row = tid >> 2;        // 0..63 edge row
    int j   = tid & 3;         // col octet
    int eid = e0 + row;
    int sN  = edge_index[eid];
    int dN  = edge_index[EE + eid];
    if (j == 0) sdstScaled[row] = (unsigned)dN * 1024u;
    unsigned short* mr = &mt[row][0];

    const float4* hd = (const float4*)(h + dN * FF + j * 8);
    const float4* hs = (const float4*)(h + sN * FF + j * 8);
    float4 d0 = hd[0], d1 = hd[1];
    float4 s0 = hs[0], s1 = hs[1];
    uint4 p;
    p.x = pk2(d0.x, d0.y); p.y = pk2(d0.z, d0.w);
    p.z = pk2(d1.x, d1.y); p.w = pk2(d1.z, d1.w);
    *(uint4*)(mr + j * 8) = p;
    p.x = pk2(s0.x, s0.y); p.y = pk2(s0.z, s0.w);
    p.z = pk2(s1.x, s1.y); p.w = pk2(s1.z, s1.w);
    *(uint4*)(mr + 32 + j * 8) = p;

    // e = ea @ Wee + bee  (8 output cols)
    float ac[8];
#pragma unroll
    for (int k = 0; k < 8; ++k) ac[k] = beel[j * 8 + k];
    const float* ear = edge_emb + edge_attr[eid] * FEE;
#pragma unroll
    for (int k2 = 0; k2 < FEE; ++k2) {
      float av = ear[k2];
      const float* wr = Weel + k2 * FF + j * 8;
#pragma unroll
      for (int k = 0; k < 8; ++k) ac[k] += av * wr[k];
    }
    p.x = pk2(ac[0], ac[1]); p.y = pk2(ac[2], ac[3]);
    p.z = pk2(ac[4], ac[5]); p.w = pk2(ac[6], ac[7]);
    *(uint4*)(mr + 64 + j * 8) = p;
  }
  __syncthreads();

  // ---- stage B: WMMA sweep ----
  const int wave = tid >> 5, lane = tid & 31;
  const int nlow = lane & 15, hi = lane >> 4;

  union AB { uint4 u[2]; v16bf v; };
  float* aggF = (float*)agg4;

#pragma unroll
  for (int sub = 0; sub < 2; ++sub) {
    int ntile = wave * 2 + sub;
    // load B fragments once, reuse over 4 M-tiles
    AB bb[3];
#pragma unroll
    for (int kc = 0; kc < 3; ++kc) {
      const unsigned short* bp = Wpk + (size_t)(ntile * 3 + kc) * 512 + lane * 16;
      bb[kc].u[0] = *(const uint4*)bp;
      bb[kc].u[1] = *(const uint4*)(bp + 8);
    }
    int col = ntile * 16 + nlow;
    float bias = bprel[col];
    unsigned col4 = (unsigned)col * 4u;

#pragma unroll
    for (int mt_i = 0; mt_i < 4; ++mt_i) {
      // A fragments per ISA 16-bit A 16x32 layout: two contiguous 16B runs per K-chunk
      const unsigned short* mr = &mt[mt_i * 16 + nlow][0];
      AB afrag[3];
#pragma unroll
      for (int kc = 0; kc < 3; ++kc) {
        afrag[kc].u[0] = *(const uint4*)(mr + kc * 32 + 8 * hi);
        afrag[kc].u[1] = *(const uint4*)(mr + kc * 32 + 16 + 8 * hi);
      }
      v8f acc = {};
#pragma unroll
      for (int kc = 0; kc < 3; ++kc) {
        acc = __builtin_amdgcn_wmma_f32_16x16x32_bf16(
            false, afrag[kc].v, false, bb[kc].v, (short)0, acc, false, false);
      }
      // ---- bias + segment atomics (C/D layout: row = v + 8*hi, col = ntile*16 + lane%16)
#pragma unroll
      for (int v = 0; v < 8; ++v) {
        int erow = mt_i * 16 + v + 8 * hi;
        float val = acc[v] + bias;
        size_t base = (size_t)(sdstScaled[erow] + col4);   // single 32-bit add, then widen
        unsigned enc = encf(val);
        atomicAdd(aggF + base, val);
        atomicAdd(aggF + base + 1, val * val);
        atomicMin(agg4 + base + 2, enc);
        atomicMax(agg4 + base + 3, enc);
      }
    }
  }
}

// ---------------- per-node aggregates + post-MLP + Wlin + BN stats ----------------
__global__ __launch_bounds__(256) void node_post_kernel(
    const uint4* __restrict__ agg4,
    const float* __restrict__ deg, const float* __restrict__ ampA, const float* __restrict__ attA,
    const float* __restrict__ h, const float* __restrict__ Wpostl, const float* __restrict__ bpostl,
    const float* __restrict__ Wlinl, const float* __restrict__ blinl,
    float* __restrict__ outlin, float* __restrict__ bnstat) {
  __shared__ float agg[4][256];   // mean, min, max, std over (t*32+f)
  __shared__ float hn[FF];
  __shared__ float o4[TT * FOUTT];

  const int n = blockIdx.x, tid = threadIdx.x;
  float dg = deg[n];
  float ic = 1.0f / fmaxf(dg, 1.0f);
  bool has = dg > 0.0f;

  uint4 q = agg4[(size_t)n * 256 + tid];               // one b128: {sum, sq, minE, maxE}
  float s  = __builtin_bit_cast(float, q.x);
  float sq = __builtin_bit_cast(float, q.y);
  float mean = s * ic;
  float var  = sq * ic - mean * mean;
  agg[0][tid] = mean;
  agg[1][tid] = has ? decf(q.z) : 0.0f;
  agg[2][tid] = has ? decf(q.w) : 0.0f;
  agg[3][tid] = sqrtf(fmaxf(var, 0.0f) + EPSF);
  if (tid < FF) hn[tid] = h[n * FF + tid];
  __syncthreads();

  if (tid < TT * FOUTT) {
    int t = tid >> 2, fo = tid & 3;
    const float* W = Wpostl + (size_t)t * (13 * FF) * FOUTT + fo;  // stride FOUTT per input
    float acc = bpostl[t * FOUTT + fo];
#pragma unroll
    for (int c = 0; c < FF; ++c) acc += hn[c] * W[c * FOUTT];
    float am = ampA[n], at = attA[n];
    for (int a = 0; a < 4; ++a) {
#pragma unroll
      for (int f = 0; f < FF; ++f) {
        float v = agg[a][t * FF + f];
        int cb = FF + a * FF + f;
        acc += v * W[cb * FOUTT];
        acc += v * am * W[(cb + 128) * FOUTT];
        acc += v * at * W[(cb + 256) * FOUTT];
      }
    }
    o4[tid] = acc;
  }
  __syncthreads();

  if (tid < FF) {
    float acc = blinl[tid];
#pragma unroll
    for (int k = 0; k < FF; ++k) acc += o4[k] * Wlinl[k * FF + tid];
    outlin[(size_t)n * FF + tid] = acc;
    atomicAdd(&bnstat[tid], acc);
    atomicAdd(&bnstat[FF + tid], acc * acc);
  }
}

// ---------------- BN (training-mode stats) + relu ----------------
__global__ void bn_relu_kernel(const float* __restrict__ outlin, const float* __restrict__ bnstat,
                               const float* __restrict__ g, const float* __restrict__ b,
                               float* __restrict__ hout) {
  int i = blockIdx.x * blockDim.x + threadIdx.x;     // N*32
  if (i >= NN * FF) return;
  int f = i & 31;
  float mu  = bnstat[f] * (1.0f / NN);
  float var = bnstat[FF + f] * (1.0f / NN) - mu * mu;
  float sc  = g[f] * rsqrtf(var + EPSF);
  float sh  = b[f] - mu * sc;
  hout[i] = fmaxf(outlin[i] * sc + sh, 0.0f);
}

// ---------------- global add pool ----------------
__global__ void pool_kernel(const float* __restrict__ h, const int* __restrict__ batch,
                            float* __restrict__ g) {
  int i = blockIdx.x * blockDim.x + threadIdx.x;     // N*32
  if (i < NN * FF) atomicAdd(&g[batch[i >> 5] * FF + (i & 31)], h[i]);
}

// ---------------- MLP head: one block, staged in LDS ----------------
__global__ __launch_bounds__(256) void head_kernel(
    const float* __restrict__ g, const float* __restrict__ Wm0, const float* __restrict__ bm0,
    const float* __restrict__ Wmh, const float* __restrict__ bmh,
    const float* __restrict__ bnmg, const float* __restrict__ bnmb,
    const float* __restrict__ Wf, const float* __restrict__ bf, float* __restrict__ out) {
  __shared__ float A[BB][HH];
  __shared__ float sc[HH], sh[HH];
  const int tid = threadIdx.x;

  // stage 1: A = g @ Wm0 + bm0
  for (int i = tid; i < BB * HH; i += 256) {
    int r = i >> 7, cl = i & 127;
    float a = bm0[cl];
#pragma unroll
    for (int k = 0; k < FF; ++k) a += g[r * FF + k] * Wm0[k * HH + cl];
    A[r][cl] = a;
  }
  __syncthreads();

  for (int stage = 0; stage < 3; ++stage) {
    // BN over batch dim + relu, in place
    if (tid < HH) {
      float s = 0.f, q = 0.f;
      for (int r = 0; r < BB; ++r) { float v = A[r][tid]; s += v; q += v * v; }
      float mu  = s * (1.0f / BB);
      float var = q * (1.0f / BB) - mu * mu;
      float scale = bnmg[stage * HH + tid] * rsqrtf(var + EPSF);
      sc[tid] = scale;
      sh[tid] = bnmb[stage * HH + tid] - mu * scale;
    }
    __syncthreads();
    for (int i = tid; i < BB * HH; i += 256) {
      int r = i >> 7, cl = i & 127;
      A[r][cl] = fmaxf(A[r][cl] * sc[cl] + sh[cl], 0.0f);
    }
    __syncthreads();

    if (stage < 2) {
      // next = A @ Wmh[stage] + bmh[stage]   (compute to regs, then write back)
      float tmp[32];
      int j = 0;
      for (int i = tid; i < BB * HH; i += 256, ++j) {
        int r = i >> 7, cl = i & 127;
        float a = bmh[stage * HH + cl];
        const float* W = Wmh + (size_t)stage * HH * HH;
#pragma unroll 4
        for (int k = 0; k < HH; ++k) a += A[r][k] * W[k * HH + cl];
        tmp[j] = a;
      }
      __syncthreads();
      j = 0;
      for (int i = tid; i < BB * HH; i += 256, ++j) A[i >> 7][i & 127] = tmp[j];
      __syncthreads();
    }
  }

  if (tid < BB) {
    float l = bf[0];
#pragma unroll 4
    for (int k = 0; k < HH; ++k) l += A[tid][k] * Wf[k];
    out[tid]      = 1.0f / (1.0f + expf(-l));   // sigmoid
    out[BB + tid] = l;                          // logits
  }
}

// ---------------- host ----------------
extern "C" void kernel_launch(void* const* d_in, const int* in_sizes, int n_in,
                              void* d_out, int out_size, void* d_ws, size_t ws_size,
                              hipStream_t stream) {
  (void)in_sizes; (void)n_in; (void)out_size; (void)ws_size;
  const int*   x          = (const int*)d_in[0];
  const int*   edge_index = (const int*)d_in[1];
  const int*   edge_attr  = (const int*)d_in[2];
  const int*   batch      = (const int*)d_in[3];
  const float* node_emb   = (const float*)d_in[4];
  const float* edge_emb   = (const float*)d_in[5];
  const float* Wee   = (const float*)d_in[6];
  const float* bee   = (const float*)d_in[7];
  const float* Wpre  = (const float*)d_in[8];
  const float* bpre  = (const float*)d_in[9];
  const float* Wpost = (const float*)d_in[10];
  const float* bpost = (const float*)d_in[11];
  const float* Wlin  = (const float*)d_in[12];
  const float* blin  = (const float*)d_in[13];
  const float* bng   = (const float*)d_in[14];
  const float* bnb   = (const float*)d_in[15];
  const float* Wm0   = (const float*)d_in[16];
  const float* bm0   = (const float*)d_in[17];
  const float* Wmh   = (const float*)d_in[18];
  const float* bmh   = (const float*)d_in[19];
  const float* bnmg  = (const float*)d_in[20];
  const float* bnmb  = (const float*)d_in[21];
  const float* Wf    = (const float*)d_in[22];
  const float* bf    = (const float*)d_in[23];

  // workspace carve-up
  char* w = (char*)d_ws;
  size_t off = 0;
  auto carve = [&](size_t bytes) { size_t r = off; off += (bytes + 255) & ~(size_t)255; return r; };
  float* h0      = (float*)(w + carve((size_t)NN * FF * 4));
  float* h1      = (float*)(w + carve((size_t)NN * FF * 4));
  float* deg     = (float*)(w + carve((size_t)NN * 4));
  float* ampA    = (float*)(w + carve((size_t)NN * 4));
  float* attA    = (float*)(w + carve((size_t)NN * 4));
  float* avgacc  = (float*)(w + carve(256));
  unsigned short* Wpk = (unsigned short*)(w + carve((size_t)LL * 48 * 512 * 2));
  unsigned* agg4 = (unsigned*)(w + carve((size_t)NN * 256 * 16));   // interleaved {sum,sq,min,max}
  float* outlin  = (float*)(w + carve((size_t)NN * FF * 4));
  float* bnstat  = (float*)(w + carve(256));
  float* gpool   = (float*)(w + carve((size_t)BB * FF * 4));

  const long long NAGG = (long long)NN * 256;   // uint4 groups

  // degree + scalers + embedding + weight repack
  fill_u32<<<(NN + 255) / 256, 256, 0, stream>>>((unsigned*)deg, 0u, NN);
  fill_u32<<<1, 64, 0, stream>>>((unsigned*)avgacc, 0u, 64);
  fill_u32<<<(BB * FF + 255) / 256, 256, 0, stream>>>((unsigned*)gpool, 0u, BB * FF);
  embed_kernel<<<NN * FF / 256, 256, 0, stream>>>(x, node_emb, h0);
  repack_kernel<<<(LL * 48 * 512 + 255) / 256, 256, 0, stream>>>(Wpre, Wpk);
  deg_count_kernel<<<EE / 256, 256, 0, stream>>>(edge_index, deg);
  log_sum_kernel<<<NN / 256, 256, 0, stream>>>(deg, avgacc);
  scalers_kernel<<<NN / 256, 256, 0, stream>>>(deg, avgacc, ampA, attA);

  float* cur = h0;
  float* nxt = h1;
  for (int l = 0; l < LL; ++l) {
    fill_agg4<<<(unsigned)((NAGG + 255) / 256), 256, 0, stream>>>((uint4*)agg4, NAGG);

    edge_msg_wmma_kernel<<<EE / EPB, 256, 0, stream>>>(
        edge_index, edge_attr, edge_emb,
        Wee + (size_t)l * FEE * FF, bee + (size_t)l * FF, bpre + (size_t)l * TT * FF,
        cur, Wpk + (size_t)l * 48 * 512, agg4);

    fill_u32<<<1, 64, 0, stream>>>((unsigned*)bnstat, 0u, 64);
    node_post_kernel<<<NN, 256, 0, stream>>>(
        (const uint4*)agg4, deg, ampA, attA, cur,
        Wpost + (size_t)l * TT * (13 * FF) * FOUTT, bpost + (size_t)l * TT * FOUTT,
        Wlin + (size_t)l * FF * FF, blin + (size_t)l * FF, outlin, bnstat);
    bn_relu_kernel<<<NN * FF / 256, 256, 0, stream>>>(
        outlin, bnstat, bng + (size_t)l * FF, bnb + (size_t)l * FF, nxt);

    float* t = cur; cur = nxt; nxt = t;
  }

  pool_kernel<<<NN * FF / 256, 256, 0, stream>>>(cur, batch, gpool);
  head_kernel<<<1, 256, 0, stream>>>(gpool, Wm0, bm0, Wmh, bmh, bnmg, bnmb, Wf, bf,
                                     (float*)d_out);
}
